// CentroidAddressableManifold_8366596292756
// MI455X (gfx1250) — matrix operand order repaired and checked by reading, not compiled, not run on hardware
//
#include <hip/hip_runtime.h>
#include <stdint.h>

#define N_BUCKETS 512
#define SLOTS 32
#define HID 1024
#define BT 8192            // B*T = 4*2048
#define TAU 0.1f
#define ALPHA 0.5f

typedef __attribute__((ext_vector_type(16))) __bf16 v16bf;
typedef __attribute__((ext_vector_type(8)))  float  v8f;
typedef unsigned int v4u  __attribute__((ext_vector_type(4)));
typedef int          v4i  __attribute__((ext_vector_type(4)));
typedef int          v8i_t __attribute__((ext_vector_type(8)));

union BF16Vec { uint4 q[2]; v16bf v; };

#if __has_builtin(__builtin_amdgcn_tensor_load_to_lds) && __has_builtin(__builtin_amdgcn_s_wait_tensorcnt)
#define USE_TDM 1
#else
#define USE_TDM 0
#endif

__device__ __forceinline__ unsigned short f2bf(float f) {
  unsigned int u = __float_as_uint(f);
  unsigned int r = u + 0x7FFFu + ((u >> 16) & 1u);   // round-to-nearest-even
  return (unsigned short)(r >> 16);
}

// Generic LDS pointer -> 32-bit LDS byte offset (aperture: addr[31:0] is the
// LDS offset, per ISA 10.2 aperture address calculation).
__device__ __forceinline__ unsigned lds_off32(const void* p) {
  return (unsigned)(uintptr_t)p;
}

#if USE_TDM
// Issue one TDM gather: 16 rows of 2KB (bf16 uq rows) selected by 16-bit row
// indices, compacted into LDS at lds_off. D# packing per CDNA5 ISA.
__device__ __forceinline__ void issue_uq_tdm(const unsigned int* __restrict__ uq_base,
                                             unsigned lds_off,
                                             const int* __restrict__ order,
                                             int base, int tb, int cnt) {
  const uint64_t ga = (uint64_t)(uintptr_t)uq_base;

  v4u g0;
  g0[0] = 0x80000001u;                       // count=1, gather_mode=1, 16b idx
  g0[1] = lds_off;                           // lds_addr
  g0[2] = (unsigned)ga;                      // global_addr[31:0]
  g0[3] = (unsigned)((ga >> 32) & 0x01FFFFFFu) | 0x80000000u; // addr[56:32], type=2

  int rem = cnt - tb; if (rem > 16) rem = 16;

  v8i_t g1;
  g1[0] = (int)(3u << 16);                   // workgroup_mask=0, data_size=8B
  g1[1] = (int)(256u << 16);                 // tensor_dim0 = 256 (8B units) low16
  g1[2] = (int)((unsigned)BT << 16);         // tensor_dim0 hi=0 | tensor_dim1=8192 low16
  g1[3] = (int)(256u << 16);                 // tensor_dim1 hi=0 | tile_dim0=256
  g1[4] = rem & 0xFFFF;                      // tile_dim1 = #valid gather indices
  g1[5] = 256;                               // tensor_dim0_stride low32 (8B units)
  g1[6] = 0;
  g1[7] = 0;

  unsigned id[16];
  #pragma unroll
  for (int m = 0; m < 16; ++m) {
    const int row = tb + m;
    int v = (row < cnt) ? order[base + row] : 0;
    id[m] = (unsigned)__builtin_amdgcn_readfirstlane(v) & 0xFFFFu;
  }
  v4i g2, g3;
  #pragma unroll
  for (int j = 0; j < 4; ++j) {
    g2[j] = (int)(id[2 * j]     | (id[2 * j + 1] << 16));
    g3[j] = (int)(id[8 + 2 * j] | (id[8 + 2 * j + 1] << 16));
  }

#if defined(__clang_major__) && (__clang_major__ >= 23)
  v8i_t g4 = {};
  __builtin_amdgcn_tensor_load_to_lds(g0, g1, g2, g3, g4, 0);
#else
  __builtin_amdgcn_tensor_load_to_lds(g0, g1, g2, g3, 0);
#endif
}
#endif // USE_TDM

// ---------------------------------------------------------------- zero counts
__global__ void k_zero_counts(int* __restrict__ counts) {
  counts[threadIdx.x] = 0;
}

// ------------------------------------------------- unified query prep (bf16)
__global__ void k_prep_query(const float* __restrict__ q,
                             const int*   __restrict__ tids,
                             const float* __restrict__ codebook,
                             unsigned short* __restrict__ uq,   // [BT][HID] bf16
                             int*   __restrict__ counts,
                             float* __restrict__ out_buckets) {
  __shared__ float red[256];
  const int tok = blockIdx.x;
  const int t   = threadIdx.x;

  const float4 qv = ((const float4*)(q + (size_t)tok * HID))[t];
  red[t] = qv.x*qv.x + qv.y*qv.y + qv.z*qv.z + qv.w*qv.w;
  for (int off = 128; off > 0; off >>= 1) {
    __syncthreads();
    if (t < off) red[t] += red[t + off];
  }
  __syncthreads();
  const float inv1 = 1.0f / fmaxf(sqrtf(red[0]), 1e-12f);
  __syncthreads();

  const int bucket = tids[tok] % N_BUCKETS;
  const float4 av = ((const float4*)(codebook + (size_t)bucket * HID))[t];
  float4 u;
  u.x = ALPHA * qv.x * inv1 + (1.0f - ALPHA) * av.x;
  u.y = ALPHA * qv.y * inv1 + (1.0f - ALPHA) * av.y;
  u.z = ALPHA * qv.z * inv1 + (1.0f - ALPHA) * av.z;
  u.w = ALPHA * qv.w * inv1 + (1.0f - ALPHA) * av.w;

  red[t] = u.x*u.x + u.y*u.y + u.z*u.z + u.w*u.w;
  for (int off = 128; off > 0; off >>= 1) {
    __syncthreads();
    if (t < off) red[t] += red[t + off];
  }
  __syncthreads();
  const float inv2 = 1.0f / fmaxf(sqrtf(red[0]), 1e-12f);

  uint2 pk;
  pk.x = (unsigned)f2bf(u.x * inv2) | ((unsigned)f2bf(u.y * inv2) << 16);
  pk.y = (unsigned)f2bf(u.z * inv2) | ((unsigned)f2bf(u.w * inv2) << 16);
  *(uint2*)(uq + (size_t)tok * HID + t * 4) = pk;

  if (t == 0) {
    out_buckets[tok] = (float)bucket;
    atomicAdd(&counts[bucket], 1);
  }
}

// ---------------------------------------------------------------- bucket scan
__global__ void k_scan(const int* __restrict__ counts,
                       int* __restrict__ offsets,
                       int* __restrict__ cursor) {
  __shared__ int s[N_BUCKETS];
  const int t = threadIdx.x;
  s[t] = counts[t];
  for (int off = 1; off < N_BUCKETS; off <<= 1) {
    __syncthreads();
    const int v = (t >= off) ? s[t - off] : 0;
    __syncthreads();
    s[t] += v;
  }
  __syncthreads();
  offsets[t + 1] = s[t];
  const int excl = (t == 0) ? 0 : s[t - 1];
  if (t == 0) offsets[0] = 0;
  cursor[t] = excl;
}

// -------------------------------------------------------- token scatter/sort
__global__ void k_scatter(const int* __restrict__ tids,
                          int* __restrict__ cursor,
                          int* __restrict__ order) {
  const int tok = blockIdx.x * blockDim.x + threadIdx.x;
  if (tok >= BT) return;
  const int bucket = tids[tok] % N_BUCKETS;
  const int pos = atomicAdd(&cursor[bucket], 1);
  order[pos] = tok;
}

// ------------------------------------------------------------- main per-bucket
// grid = 512 (one block per bucket), 256 threads = 8 wave32 waves
__launch_bounds__(256, 1)
__global__ void k_main(const float* __restrict__ slot_keys,
                       const float* __restrict__ slot_values,
                       const int*   __restrict__ slot_tids,
                       const int*   __restrict__ tids,
                       const unsigned int* __restrict__ uqw,  // uq bf16 pairs [BT][512]
                       const int*   __restrict__ offsets,
                       const int*   __restrict__ order,
                       float* __restrict__ out_val,
                       float* __restrict__ out_max) {
  extern __shared__ __align__(16) unsigned char smem[];
  unsigned short* keysL  = (unsigned short*)smem;        // [32][1024] bf16 (64KB)
  unsigned short* valsT  = keysL + SLOTS * HID;          // [1024][32] bf16 (64KB, transposed)
  unsigned short* uqT    = valsT + SLOTS * HID;          // 2 x [16][1024] bf16 (64KB, dbl buf)
  float*          scoresL = (float*)(uqT + 2 * 16 * HID);            // [16][32] f32
  unsigned short* probsL  = (unsigned short*)(scoresL + 16 * SLOTS); // [16][32] bf16
  int*            stidsL  = (int*)(probsL + 16 * SLOTS); // [32]
  int*            toksL   = stidsL + SLOTS;              // [16]

  const int bk   = blockIdx.x;
  const int t    = threadIdx.x;
  const int base = offsets[bk];
  const int cnt  = offsets[bk + 1] - base;
  if (cnt == 0) return;

  const int wave = t >> 5;
  const int lane = t & 31;
  const int ntiles = (cnt + 15) >> 4;

  if (t < SLOTS) stidsL[t] = slot_tids[bk * SLOTS + t];

#if USE_TDM
  if (wave == 0)   // prefetch uq tile 0 while we stage the bucket tables
    issue_uq_tdm(uqw, lds_off32(uqT), order, base, 0, cnt);
#endif

  // ---- stage bucket tables into LDS with f32->bf16 conversion
  {
    const float4* kp = (const float4*)(slot_keys   + (size_t)bk * SLOTS * HID);
    const float4* vp = (const float4*)(slot_values + (size_t)bk * SLOTS * HID);
    #pragma unroll 4
    for (int i = 0; i < 32; ++i) {               // 8192 float4 / 256 thr
      const int idx = i * 256 + t;
      const int s   = idx >> 8;                  // slot
      const int d4  = idx & 255;                 // float4 within row
      if (i < 31) {                              // global_prefetch_b8 one step ahead
        __builtin_prefetch(kp + idx + 256, 0, 0);
        __builtin_prefetch(vp + idx + 256, 0, 0);
      }
      const float4 kv = kp[idx];
      const float4 vv = vp[idx];
      unsigned int* kd = (unsigned int*)(keysL + s * HID + d4 * 4);
      kd[0] = (unsigned)f2bf(kv.x) | ((unsigned)f2bf(kv.y) << 16);
      kd[1] = (unsigned)f2bf(kv.z) | ((unsigned)f2bf(kv.w) << 16);
      const int d0 = d4 * 4;
      valsT[(d0 + 0) * SLOTS + s] = f2bf(vv.x);
      valsT[(d0 + 1) * SLOTS + s] = f2bf(vv.y);
      valsT[(d0 + 2) * SLOTS + s] = f2bf(vv.z);
      valsT[(d0 + 3) * SLOTS + s] = f2bf(vv.w);
    }
  }
  __syncthreads();

  for (int tile = 0; tile < ntiles; ++tile) {
    const int tb = tile * 16;
    unsigned short* buf = uqT + (tile & 1) * 16 * HID;

    scoresL[t] = 0.f;
    scoresL[t + 256] = 0.f;

#if USE_TDM
    if (wave == 0) {
      if (tile + 1 < ntiles) {                 // prefetch next tile into other buf
        issue_uq_tdm(uqw, lds_off32(uqT + ((tile + 1) & 1) * 16 * HID),
                     order, base, tb + 16, cnt);
        __builtin_amdgcn_s_wait_tensorcnt(1);  // current buf (older op) done
      } else {
        __builtin_amdgcn_s_wait_tensorcnt(0);
      }
    }
#else
    // synchronous staging fallback (zero-pads invalid rows)
    for (int i = 0; i < 32; ++i) {             // 8192 uints
      const int idx = i * 256 + t;
      const int m = idx >> 9;
      const int j = idx & 511;
      const int row = tb + m;
      unsigned int v = 0;
      if (row < cnt) v = uqw[(size_t)order[base + row] * 512 + j];
      ((unsigned int*)buf)[idx] = v;
    }
#endif
    __syncthreads();

    // ---- scores[16 tok x 32 slot] = uq(16xK) * keys^T(Kx16), K=1024
    // 8 waves = 2 slot-tiles x 4 K-partitions (256 K each = 8 WMMAs)
    {
      const int st   = wave & 1;
      const int kp4  = wave >> 1;
      const int mrow = lane & 15;
      const int sel  = (lane >> 4) << 3;       // A: 0 or 8
      const int slot = st * 16 + (lane & 15);
      const int kb   = (lane >> 4) << 3;       // B: 0 or 8
      v8f acc = {};
      #pragma unroll
      for (int kk = 0; kk < 8; ++kk) {
        const int kc = kp4 * 256 + kk * 32;
        BF16Vec a, b;
        a.q[0] = *(const uint4*)(buf + mrow * HID + kc + sel);
        a.q[1] = *(const uint4*)(buf + mrow * HID + kc + 16 + sel);
        const uint4* bp = (const uint4*)(keysL + slot * HID + kc + kb);
        b.q[0] = bp[0];
        b.q[1] = bp[2];
        acc = __builtin_amdgcn_wmma_f32_16x16x32_bf16(
            false, a.v, false, b.v, (short)0, acc, false, false);
      }
      const int mb = (lane >> 4) << 3;
      #pragma unroll
      for (int r = 0; r < 8; ++r)              // merge K partials: ds_add_f32
        atomicAdd(&scoresL[(r + mb) * SLOTS + st * 16 + (lane & 15)], acc[r]);
    }
    __syncthreads();

    // ---- hard-match / softmax weights, max_sim (threads 0..15)
    if (t < 16) {
      const int row = tb + t;
      if (row < cnt) {
        const int tok = order[base + row];
        toksL[t] = tok;
        const int mytid = tids[tok];
        float maxs = -1e30f;
        int msum = 0;
        #pragma unroll
        for (int s = 0; s < SLOTS; ++s) {
          maxs = fmaxf(maxs, scoresL[t * SLOTS + s]);
          msum += (stidsL[s] == mytid) ? 1 : 0;
        }
        if (msum > 0) {
          const float p = 1.0f / ((float)msum + 1e-9f);
          const unsigned short pb = f2bf(p);
          #pragma unroll
          for (int s = 0; s < SLOTS; ++s)
            probsL[t * SLOTS + s] = (stidsL[s] == mytid) ? pb : (unsigned short)0;
          out_max[tok] = 10.0f;
        } else {
          float e[SLOTS];
          float sum = 0.f;
          #pragma unroll
          for (int s = 0; s < SLOTS; ++s) {
            e[s] = __expf((scoresL[t * SLOTS + s] - maxs) * (1.0f / TAU));
            sum += e[s];
          }
          const float inv = 1.0f / sum;
          #pragma unroll
          for (int s = 0; s < SLOTS; ++s) probsL[t * SLOTS + s] = f2bf(e[s] * inv);
          out_max[tok] = maxs;
        }
      } else {
        toksL[t] = -1;
        #pragma unroll
        for (int s = 0; s < SLOTS; ++s) probsL[t * SLOTS + s] = 0;
      }
    }
    __syncthreads();

    // ---- out(16x1024) = probs(16x32) * vals(32x1024); one WMMA per 16-col tile
    {
      const int mrow = lane & 15;
      const int sel  = (lane >> 4) << 3;
      BF16Vec a;
      a.q[0] = *(const uint4*)(probsL + mrow * SLOTS + sel);
      a.q[1] = *(const uint4*)(probsL + mrow * SLOTS + 16 + sel);
      const int ncol = lane & 15;
      const int kb   = (lane >> 4) << 3;
      const int mb   = (lane >> 4) << 3;
      for (int nt = wave; nt < HID / 16; nt += 8) {
        BF16Vec b;
        const uint4* bp = (const uint4*)(valsT + (nt * 16 + ncol) * SLOTS + kb);
        b.q[0] = bp[0];
        b.q[1] = bp[2];
        v8f c = {};
        v8f d = __builtin_amdgcn_wmma_f32_16x16x32_bf16(
            false, a.v, false, b.v, (short)0, c, false, false);
        #pragma unroll
        for (int r = 0; r < 8; ++r) {
          const int tok = toksL[r + mb];
          if (tok >= 0)
            out_val[(size_t)tok * HID + nt * 16 + ncol] = d[r];
        }
      }
    }
    __syncthreads();
  }
}

// ---------------------------------------------------------------------- launch
extern "C" void kernel_launch(void* const* d_in, const int* in_sizes, int n_in,
                              void* d_out, int out_size, void* d_ws, size_t ws_size,
                              hipStream_t stream) {
  const float* q        = (const float*)d_in[0];   // [4,2048,1024]
  const int*   tids     = (const int*)d_in[1];     // [4,2048]
  const float* skeys    = (const float*)d_in[2];   // [16384,1024]
  const float* svals    = (const float*)d_in[3];   // [16384,1024]
  const int*   stids    = (const int*)d_in[4];     // [16384]
  const float* codebook = (const float*)d_in[5];   // [512,1024]

  float* out         = (float*)d_out;
  float* out_val     = out;                        // 8192*1024
  float* out_max     = out + (size_t)BT * HID;     // 8192
  float* out_buckets = out_max + BT;               // 8192

  unsigned char* ws = (unsigned char*)d_ws;
  unsigned short* uq = (unsigned short*)ws;        // BT*HID bf16 = 16MB
  size_t off = (size_t)BT * HID * sizeof(unsigned short);
  int* counts  = (int*)(ws + off); off += N_BUCKETS * sizeof(int);
  int* offsets = (int*)(ws + off); off += (N_BUCKETS + 1) * sizeof(int);
  int* cursor  = (int*)(ws + off); off += N_BUCKETS * sizeof(int);
  int* order   = (int*)(ws + off); off += BT * sizeof(int);

  k_zero_counts<<<1, N_BUCKETS, 0, stream>>>(counts);
  k_prep_query<<<BT, 256, 0, stream>>>(q, tids, codebook, uq, counts, out_buckets);
  k_scan<<<1, N_BUCKETS, 0, stream>>>(counts, offsets, cursor);
  k_scatter<<<BT / 256, 256, 0, stream>>>(tids, cursor, order);

  const size_t smem = (size_t)(SLOTS * HID) * 2      // keysL
                    + (size_t)(SLOTS * HID) * 2      // valsT
                    + (size_t)(2 * 16 * HID) * 2     // uqT double buffer
                    + 16 * SLOTS * 4                 // scoresL
                    + 16 * SLOTS * 2                 // probsL
                    + SLOTS * 4 + 16 * 4;            // stidsL + toksL
  hipFuncSetAttribute(reinterpret_cast<const void*>(k_main),
                      hipFuncAttributeMaxDynamicSharedMemorySize, (int)smem);
  k_main<<<N_BUCKETS, 256, smem, stream>>>(skeys, svals, stids, tids,
                                           (const unsigned int*)uq,
                                           offsets, order, out_val, out_max);
}
